// RelationalTransformerUpdate_76132590289344
// MI455X (gfx1250) — matrix operand order repaired and checked by reading, not compiled, not run on hardware
//
#include <hip/hip_runtime.h>

// Problem constants (match reference)
#define BB   2
#define SS   256
#define DD   1024
#define HH   8
#define DKK  128
#define FF   4096
#define LL   8
#define KREL 37
#define EPSF 1e-5f

typedef __attribute__((ext_vector_type(16))) __bf16 v16bf;
typedef __attribute__((ext_vector_type(8)))  __bf16 v8bf;
typedef __attribute__((ext_vector_type(8)))  float  v8f;

#define LDK 72   // padded LDS row length in shorts: 144 B = 9 * 16 B (aligned, bank-spread)

__device__ __forceinline__ unsigned bf16bits(float x) {
    union { float f; unsigned u; } a; a.f = x;
    return (a.u + 0x7fffu + ((a.u >> 16) & 1u)) >> 16;   // round-nearest-even
}
__device__ __forceinline__ unsigned pack2bf(float lo, float hi) {
    return (bf16bits(lo) & 0xffffu) | (bf16bits(hi) << 16);
}

// ---------------------------------------------------------------------------
// WMMA GEMM: C[M,N] = A(M x Kd) * B(Kd x N) (+bias +residual, relu)
// Block: 256 threads (8 waves). Block tile 128x128, wave tile 32x64
// (2x4 WMMA accumulators). TK=64 per LDS stage -> 16 v_wmma per barrier pair.
// MODE 0: plain GEMM (row-major A and B)
// MODE 1: attention scores  Q*K^T per (b,h)   (blockIdx.z = b*H+h)
// MODE 2: attention output  P*V   per (b,h)
// MODE 3: fused QKV         (blockIdx.z selects {W,bias,out} triple)
// B element address = Bm[kk*ldbk + n*ldbn]; B staged TRANSPOSED in LDS
// (each lane's fragment = 16 contiguous bf16 -> 2x ds_load_b128), staged
// with packed uint4 stores (4x ds_store_b128 per thread per tile).
// ---------------------------------------------------------------------------
template<int MODE, bool HAS_BIAS, bool HAS_RES, bool RELU>
__global__ __launch_bounds__(256) void wmma_gemm(
    const float* __restrict__ A, const float* Bm,
    const float* bias, const float* __restrict__ Csrc,
    float* Cdst, int N, int Kd,
    int lda, int ldbk, int ldbn, int ldc,
    const float* Bm2 = nullptr, const float* bias2 = nullptr, float* Cd2 = nullptr,
    const float* Bm3 = nullptr, const float* bias3 = nullptr, float* Cd3 = nullptr)
{
    __shared__ __bf16 lA[128][LDK];    // 18 KB
    __shared__ __bf16 lBt[128][LDK];   // 18 KB  (transposed: [n][k])

    const int tid  = threadIdx.x;
    const int lane = tid & 31;
    const int wid  = tid >> 5;
    const int wm   = wid >> 1;         // 0..3  (32-row strip)
    const int wn   = wid & 1;          // 0..1  (64-col strip)

    if constexpr (MODE == 1) {         // scores = Q * K^T
        int bz = blockIdx.z, b = bz >> 3, hh = bz & 7;
        A    += (size_t)b * SS * DD + (size_t)hh * DKK;
        Bm   += (size_t)b * SS * DD + (size_t)hh * DKK;
        Cdst += (size_t)bz * SS * SS;
    } else if constexpr (MODE == 2) {  // o = P * V
        int bz = blockIdx.z, b = bz >> 3, hh = bz & 7;
        A    += (size_t)bz * SS * SS;
        Bm   += (size_t)b * SS * DD + (size_t)hh * DKK;
        Cdst += (size_t)b * SS * DD + (size_t)hh * DKK;
    } else if constexpr (MODE == 3) {  // fused QKV: z picks weight/bias/out
        int z = blockIdx.z;
        if (z == 1)      { Bm = Bm2; bias = bias2; Cdst = Cd2; }
        else if (z == 2) { Bm = Bm3; bias = bias3; Cdst = Cd3; }
    }

    const int rowBase = blockIdx.y * 128;
    const int colBase = blockIdx.x * 128;

    // Staging coordinates (fixed per thread)
    const int sar = tid >> 1;               // A row (0..127), 32 K per thread
    const int sac = (tid & 1) << 5;         // A k-col: 0 or 32
    const int scol = tid & 127;             // B column owned by this thread
    const int sk0  = (tid >> 7) << 5;       // B k-base: 0 or 32

    const int g8  = (lane >> 4) << 3;       // unused; A and B frags both use kb (0/16)
    const int kb  = (lane >> 4) << 4;       // B frag K-half select (0/16)
    const int m15 = lane & 15;

    v8f acc[2][4] = {};

    for (int kc = 0; kc < Kd; kc += 64) {
        // Prefetch next K-tile of the streams (global_prefetch_b8)
        if (kc + 64 < Kd) {
            __builtin_prefetch(A + (size_t)(rowBase + sar) * lda + kc + 64 + sac, 0, 1);
            __builtin_prefetch(Bm + (size_t)(kc + 64 + sk0) * ldbk
                                  + (size_t)(colBase + scol) * ldbn, 0, 1);
        }

        // Stage A: 32 consecutive K in one row; pack -> 4x uint4 (ds_store_b128)
        {
            const float* src = A + (size_t)(rowBase + sar) * lda + kc + sac;
            unsigned tmp[16];
            #pragma unroll
            for (int i = 0; i < 16; ++i)
                tmp[i] = pack2bf(src[2 * i], src[2 * i + 1]);
            uint4* dst = reinterpret_cast<uint4*>(&lA[sar][sac]);
            #pragma unroll
            for (int w = 0; w < 4; ++w)
                dst[w] = make_uint4(tmp[4*w], tmp[4*w+1], tmp[4*w+2], tmp[4*w+3]);
        }
        // Stage B transposed: one column, 32 K values; pack -> 4x uint4
        {
            const float* src = Bm + (size_t)(kc + sk0) * ldbk
                                  + (size_t)(colBase + scol) * ldbn;
            unsigned tmp[16];
            #pragma unroll
            for (int i = 0; i < 16; ++i)
                tmp[i] = pack2bf(src[(size_t)(2 * i) * ldbk],
                                 src[(size_t)(2 * i + 1) * ldbk]);
            uint4* dst = reinterpret_cast<uint4*>(&lBt[scol][sk0]);
            #pragma unroll
            for (int w = 0; w < 4; ++w)
                dst[w] = make_uint4(tmp[4*w], tmp[4*w+1], tmp[4*w+2], tmp[4*w+3]);
        }
        __syncthreads();

        // Two K=32 WMMA chunks per stage; 8 wmma each
        #pragma unroll
        for (int c = 0; c < 64; c += 32) {
            v16bf af[2], bfrag[4];
            #pragma unroll
            for (int i = 0; i < 2; ++i) {
                int row = wm * 32 + i * 16 + m15;
                v8bf lo = *(const v8bf*)&lA[row][c + kb];        // K = c+kb ..
                v8bf hi = *(const v8bf*)&lA[row][c + kb + 8];    // K = c+kb+8 ..
                af[i] = __builtin_shufflevector(lo, hi,
                        0, 1, 2, 3, 4, 5, 6, 7, 8, 9, 10, 11, 12, 13, 14, 15);
            }
            #pragma unroll
            for (int j = 0; j < 4; ++j) {
                int col = wn * 64 + j * 16 + m15;
                bfrag[j] = *(const v16bf*)&lBt[col][c + kb];     // 16 contiguous K
            }
            #pragma unroll
            for (int i = 0; i < 2; ++i)
                #pragma unroll
                for (int j = 0; j < 4; ++j)
                    acc[i][j] = __builtin_amdgcn_wmma_f32_16x16x32_bf16(
                        false, af[i], false, bfrag[j], (short)0, acc[i][j],
                        false, false);
        }
        __syncthreads();
    }

    // Epilogue: C/D layout — VGPR r: M = r + 8*(lane>=16), N = lane&15
    const int mrow8 = (lane >> 4) << 3;
    #pragma unroll
    for (int i = 0; i < 2; ++i)
        #pragma unroll
        for (int j = 0; j < 4; ++j)
            #pragma unroll
            for (int r = 0; r < 8; ++r) {
                int row = rowBase + wm * 32 + i * 16 + mrow8 + r;
                int col = colBase + wn * 64 + j * 16 + m15;
                float vv = acc[i][j][r];
                if constexpr (HAS_BIAS) vv += bias[col];
                if constexpr (HAS_RES)  vv += Csrc[(size_t)row * ldc + col];
                if constexpr (RELU)     vv  = fmaxf(vv, 0.f);
                Cdst[(size_t)row * ldc + col] = vv;
            }
}

// ---------------------------------------------------------------------------
// LayerNorm over last dim (D=1024), one row per block
// ---------------------------------------------------------------------------
__global__ __launch_bounds__(256) void ln_kernel(
    const float* __restrict__ in, const float* __restrict__ g,
    const float* __restrict__ b, float* __restrict__ out)
{
    __shared__ float rs[256], rq[256];
    const int row = blockIdx.x, tid = threadIdx.x;
    const float* x = in + (size_t)row * DD;
    float s = 0.f, s2 = 0.f;
    for (int i = tid; i < DD; i += 256) { float v = x[i]; s += v; s2 += v * v; }
    rs[tid] = s; rq[tid] = s2; __syncthreads();
    for (int st = 128; st > 0; st >>= 1) {
        if (tid < st) { rs[tid] += rs[tid + st]; rq[tid] += rq[tid + st]; }
        __syncthreads();
    }
    float mean = rs[0] * (1.f / DD);
    float var  = rq[0] * (1.f / DD) - mean * mean;
    float rstd = rsqrtf(var + EPSF);
    float* o = out + (size_t)row * DD;
    for (int i = tid; i < DD; i += 256)
        o[i] = (x[i] - mean) * rstd * g[i] + b[i];
}

// ---------------------------------------------------------------------------
// scores[b,h,q,k] = (scores + dot(q[b,h,q,:], rel_k[relation[b,q,k]])) * scale
// then masked. One thread per (b,q,k); 8 heads share the gathered row.
// ---------------------------------------------------------------------------
__global__ __launch_bounds__(256) void rel_scores_kernel(
    const float* __restrict__ q, const float* __restrict__ rk,
    const int* __restrict__ relation, const unsigned char* __restrict__ mask,
    float* __restrict__ scores)
{
    int t  = blockIdx.x * 256 + threadIdx.x;   // b*S*S + qp*S + kp
    int kp = t & (SS - 1);
    int qp = (t >> 8) & (SS - 1);
    int b  = t >> 16;
    const float* rrow = rk + (size_t)relation[t] * DKK;
    const float* qb   = q + (size_t)(b * SS + qp) * DD;
    float acc[HH] = {};
    for (int d = 0; d < DKK; ++d) {
        float rv = rrow[d];
        #pragma unroll
        for (int h = 0; h < HH; ++h) acc[h] += qb[h * DKK + d] * rv;
    }
    const float scale  = 0.088388347648318447f;   // 1/sqrt(128)
    const bool  masked = (mask[b * SS + kp] == 0);
    #pragma unroll
    for (int h = 0; h < HH; ++h) {
        size_t idx = ((size_t)(b * HH + h) * SS + qp) * SS + kp;
        float sc = (scores[idx] + acc[h]) * scale;
        if (masked) sc = -1e9f;
        scores[idx] = sc;
    }
}

// ---------------------------------------------------------------------------
// Row softmax over S=256, one row per block
// ---------------------------------------------------------------------------
__global__ __launch_bounds__(256) void softmax_kernel(float* __restrict__ scores)
{
    __shared__ float red[256];
    const int row = blockIdx.x, tid = threadIdx.x;
    float* s = scores + (size_t)row * SS;
    float v = s[tid];
    red[tid] = v; __syncthreads();
    for (int st = 128; st > 0; st >>= 1) {
        if (tid < st) red[tid] = fmaxf(red[tid], red[tid + st]);
        __syncthreads();
    }
    float mx = red[0]; __syncthreads();
    float e = __expf(v - mx);
    red[tid] = e; __syncthreads();
    for (int st = 128; st > 0; st >>= 1) {
        if (tid < st) red[tid] += red[tid + st];
        __syncthreads();
    }
    s[tid] = e / red[0];
}

// ---------------------------------------------------------------------------
// o[b,q,h*DK+d] += sum_k p[b,h,q,k] * rel_v[relation[b,q,k]][d]
// One thread per (b,q,d); gathered rel_v element shared across 8 heads.
// ---------------------------------------------------------------------------
__global__ __launch_bounds__(256) void rel_out_kernel(
    const float* __restrict__ probs, const float* __restrict__ rvw,
    const int* __restrict__ relation, float* __restrict__ o)
{
    int t  = blockIdx.x * 256 + threadIdx.x;   // b*S*DK + qp*DK + d
    int d  = t & (DKK - 1);
    int qp = (t >> 7) & (SS - 1);
    int b  = t >> 15;
    const int* rrow = relation + (size_t)(b * SS + qp) * SS;
    float acc[HH] = {};
    for (int k = 0; k < SS; ++k) {
        float rv = rvw[(size_t)rrow[k] * DKK + d];
        #pragma unroll
        for (int h = 0; h < HH; ++h)
            acc[h] += probs[((size_t)(b * HH + h) * SS + qp) * SS + k] * rv;
    }
    float* ob = o + (size_t)(b * SS + qp) * DD + d;
    #pragma unroll
    for (int h = 0; h < HH; ++h) ob[h * DKK] += acc[h];
}

// ---------------------------------------------------------------------------
extern "C" void kernel_launch(void* const* d_in, const int* in_sizes, int n_in,
                              void* d_out, int out_size, void* d_ws, size_t ws_size,
                              hipStream_t stream)
{
    (void)in_sizes; (void)n_in; (void)out_size; (void)ws_size;

    const float*         x        = (const float*)d_in[0];
    const int*           relation = (const int*)d_in[1];
    const unsigned char* mask     = (const unsigned char*)d_in[2];
    const float* Wq    = (const float*)d_in[3];
    const float* bq    = (const float*)d_in[4];
    const float* Wk    = (const float*)d_in[5];
    const float* bk    = (const float*)d_in[6];
    const float* Wv    = (const float*)d_in[7];
    const float* bv    = (const float*)d_in[8];
    const float* Wo    = (const float*)d_in[9];
    const float* bo    = (const float*)d_in[10];
    const float* ln1_g = (const float*)d_in[11];
    const float* ln1_b = (const float*)d_in[12];
    const float* W1    = (const float*)d_in[13];
    const float* b1    = (const float*)d_in[14];
    const float* W2    = (const float*)d_in[15];
    const float* b2    = (const float*)d_in[16];
    const float* ln2_g = (const float*)d_in[17];
    const float* ln2_b = (const float*)d_in[18];
    const float* rel_k = (const float*)d_in[19];
    const float* rel_v = (const float*)d_in[20];
    const float* lnf_g = (const float*)d_in[21];
    const float* lnf_b = (const float*)d_in[22];

    // Workspace layout (floats): ~24 MB total
    float* ws = (float*)d_ws;
    const size_t nHid = (size_t)BB * SS * DD;          // 524288
    float* h  = ws;
    float* hn = h  + nHid;
    float* q  = hn + nHid;
    float* k  = q  + nHid;
    float* v  = k  + nHid;
    float* o  = v  + nHid;
    float* sc = o  + nHid;                              // B*H*S*S = 1048576
    float* u  = sc + (size_t)BB * HH * SS * SS;         // B*S*F   = 2097152

    hipMemcpyAsync(h, x, nHid * sizeof(float), hipMemcpyDeviceToDevice, stream);

    const int nRows = BB * SS;                         // 512
    const dim3 gQKV (DD / 128, nRows / 128, 3);        // fused q,k,v
    const dim3 gProj(DD / 128, nRows / 128, 1);        // N=1024
    const dim3 gFfn1(FF / 128, nRows / 128, 1);        // N=4096
    const dim3 gScore(SS / 128, SS / 128, BB * HH);
    const dim3 gPV  (DKK / 128, SS / 128, BB * HH);

    for (int l = 0; l < LL; ++l) {
        const float* wq  = Wq + (size_t)l * DD * DD;
        const float* wk  = Wk + (size_t)l * DD * DD;
        const float* wv  = Wv + (size_t)l * DD * DD;
        const float* wo  = Wo + (size_t)l * DD * DD;
        const float* w1  = W1 + (size_t)l * DD * FF;
        const float* w2  = W2 + (size_t)l * FF * DD;
        const float* bq_ = bq + (size_t)l * DD;
        const float* bk_ = bk + (size_t)l * DD;
        const float* bv_ = bv + (size_t)l * DD;
        const float* bo_ = bo + (size_t)l * DD;
        const float* b1_ = b1 + (size_t)l * FF;
        const float* b2_ = b2 + (size_t)l * DD;
        const float* rkl = rel_k + (size_t)l * KREL * DKK;
        const float* rvl = rel_v + (size_t)l * KREL * DKK;

        // h2 = LN1(h)
        ln_kernel<<<nRows, 256, 0, stream>>>(h, ln1_g + (size_t)l * DD,
                                             ln1_b + (size_t)l * DD, hn);
        // fused q,k,v projections (bias only)
        wmma_gemm<3, true, false, false><<<gQKV, 256, 0, stream>>>(
            hn, wq, bq_, nullptr, q, DD, DD, DD, DD, 1, DD,
            wk, bk_, k, wv, bv_, v);

        // scores = Q*K^T per (b,h)
        wmma_gemm<1, false, false, false><<<gScore, 256, 0, stream>>>(
            q, k, nullptr, nullptr, sc, SS, DKK, DD, 1, DD, SS);
        // + relational term, scale, mask
        rel_scores_kernel<<<(BB * SS * SS) / 256, 256, 0, stream>>>(
            q, rkl, relation, mask, sc);
        // softmax rows
        softmax_kernel<<<BB * HH * SS, 256, 0, stream>>>(sc);
        // o = P*V per (b,h)
        wmma_gemm<2, false, false, false><<<gPV, 256, 0, stream>>>(
            sc, v, nullptr, nullptr, o, DKK, SS, SS, DD, 1, DD);
        // o += P*rel_v (gathered)
        rel_out_kernel<<<(BB * SS * DKK) / 256, 256, 0, stream>>>(
            sc, rvl, relation, o);

        // h = h + o @ Wo + bo
        wmma_gemm<0, true, true, false><<<gProj, 256, 0, stream>>>(
            o, wo, bo_, h, h, DD, DD, DD, DD, 1, DD);
        // h2 = LN2(h)
        ln_kernel<<<nRows, 256, 0, stream>>>(h, ln2_g + (size_t)l * DD,
                                             ln2_b + (size_t)l * DD, hn);
        // u = relu(h2 @ W1 + b1)
        wmma_gemm<0, true, false, true><<<gFfn1, 256, 0, stream>>>(
            hn, w1, b1_, nullptr, u, FF, DD, DD, FF, 1, FF);
        // h = h + u @ W2 + b2
        wmma_gemm<0, true, true, false><<<gProj, 256, 0, stream>>>(
            u, w2, b2_, h, h, DD, FF, FF, DD, 1, DD);
    }

    // out = LN_f(h)
    ln_kernel<<<nRows, 256, 0, stream>>>(h, lnf_g, lnf_b, (float*)d_out);
}
